// MultiHeadAttention_59785944760699
// MI455X (gfx1250) — compile-verified, matrix operand order checked by reading, not run on hardware
//
#include <hip/hip_runtime.h>
#include <hip/hip_bf16.h>

// ---------------------------------------------------------------------------
// MI455X (gfx1250) multi-head attention, bf16 WMMA pipeline.
//   x[4,2048,1024] fp32 -> bf16
//   Q = x@Wq^T+bq ; K = x@Wk^T+bk ; V = x@Wv^T+bv (stored transposed per head)
//   flash-attention (causal + pad mask) -> O bf16
//   out = O@Wo^T + bo (fp32)
// All matrix math uses v_wmma_f32_16x16x32_bf16 (wave32).
// GEMM: double-buffered LDS, k-step 64, one barrier per step; global->LDS
// staging uses GLOBAL_LOAD_ASYNC_TO_LDS_B128 (ASYNCcnt) when available.
// ---------------------------------------------------------------------------

typedef __bf16 bf16_t;
typedef __attribute__((ext_vector_type(8)))  __bf16 v8bf;
typedef __attribute__((ext_vector_type(16))) __bf16 v16bf;
typedef __attribute__((ext_vector_type(8)))  float  v8f;
typedef int vi4 __attribute__((vector_size(4 * sizeof(int))));

#define BATCH 4
#define SEQ   2048
#define HEADS 16
#define DHEAD 64
#define DMODEL 1024
#define MTOT  (BATCH * SEQ)   // 8192

#if defined(__has_builtin)
#if __has_builtin(__builtin_amdgcn_global_load_async_to_lds_b128)
#define USE_ASYNC_LDS 1
#endif
#endif

#define AS1 __attribute__((address_space(1)))
#define AS3 __attribute__((address_space(3)))

static __device__ inline v16bf mk16(v8bf lo, v8bf hi) {
  return __builtin_shufflevector(lo, hi, 0,1,2,3,4,5,6,7,8,9,10,11,12,13,14,15);
}

static __device__ inline v8f wmma_bf16(v16bf a, v16bf b, v8f c) {
  return __builtin_amdgcn_wmma_f32_16x16x32_bf16(false, a, false, b, (short)0, c,
                                                 false, false);
}

// 16-byte global -> LDS copy.  Async DMA path on gfx1250 when the builtin is
// declared; otherwise a plain load/store pair (still correct with the
// single-barrier double-buffer scheme).
static __device__ inline void copy16_g2l(const bf16_t* g, bf16_t* l) {
#ifdef USE_ASYNC_LDS
  // generic LDS pointer low 32 bits == LDS byte offset (ISA 10.2 aperture);
  // generic global pointer is bit-identical to AS1.  Integer casts avoid
  // illegal addrspacecasts.  Builtin params are AS1/AS3 pointers to 16-byte
  // int vectors (from the compiler diagnostic).
  __builtin_amdgcn_global_load_async_to_lds_b128(
      (AS1 vi4*)(uintptr_t)g,
      (AS3 vi4*)(unsigned int)(uintptr_t)l, 0, 0);
#else
  *(v8bf*)l = *(const v8bf*)g;
#endif
}

static __device__ inline void wait_async_copies() {
#ifdef USE_ASYNC_LDS
#if defined(__has_builtin) && __has_builtin(__builtin_amdgcn_s_wait_asynccnt)
  __builtin_amdgcn_s_wait_asynccnt(0);
#else
  asm volatile("s_wait_asynccnt 0x0" ::: "memory");
#endif
#endif
}

// ---------------------------------------------------------------------------
// fp32 -> bf16 elementwise convert (vectorized x4)
// ---------------------------------------------------------------------------
__global__ __launch_bounds__(256) void cvt_bf16_kernel(const float* __restrict__ in,
                                                       bf16_t* __restrict__ out,
                                                       int n4) {
  int i = blockIdx.x * blockDim.x + threadIdx.x;
  int stride = gridDim.x * blockDim.x;
  for (; i < n4; i += stride) {
    float4 v = ((const float4*)in)[i];
    bf16_t r[4] = {(bf16_t)v.x, (bf16_t)v.y, (bf16_t)v.z, (bf16_t)v.w};
    *(uint2*)(out + 4 * (size_t)i) = *(uint2*)r;
  }
}

// ---------------------------------------------------------------------------
// GEMM: C[M=8192, N=1024] = A[M,1024] @ W[N,1024]^T + bias
//   A, W bf16 K-contiguous.  Block tile 128x128, 4 waves, wave tile 64x64.
//   Double-buffered LDS, k-step 64 (2 WMMA k-chunks), 1 barrier / step.
//   mode 0: out bf16 row-major [M,N]          (Q, K)
//   mode 1: out bf16 transposed Vt[B,H,Dh,S]  (V)
//   mode 2: out fp32 row-major [M,N]          (final projection)
// ---------------------------------------------------------------------------
__global__ __launch_bounds__(128) void gemm_kernel(const bf16_t* __restrict__ A,
                                                   const bf16_t* __restrict__ W,
                                                   const float* __restrict__ bias,
                                                   bf16_t* __restrict__ outB,
                                                   float* __restrict__ outF,
                                                   int mode) {
  __shared__ bf16_t As[2 * 128 * 64];  // 32 KB
  __shared__ bf16_t Ws[2 * 128 * 64];  // 32 KB

  const int t    = threadIdx.x;
  const int lane = t & 31;
  const int wave = t >> 5;
  const int ln   = lane & 15;
  const int half = lane >> 4;
  const int koff = half * 8;
  const int wm = wave >> 1, wn = wave & 1;
  const int mBase = blockIdx.y * 128;
  const int nBase = blockIdx.x * 128;

  const bf16_t* arow = A + (size_t)(mBase + t) * DMODEL;
  const bf16_t* wrow = W + (size_t)(nBase + t) * DMODEL;

  v8f acc[4][4];
#pragma unroll
  for (int mt = 0; mt < 4; ++mt)
#pragma unroll
    for (int nt = 0; nt < 4; ++nt) acc[mt][nt] = (v8f)0.0f;

  // preload k-step 0 into buffer 0 (each thread owns one 64-elem row chunk)
  {
    bf16_t* as = As + t * 64;
    bf16_t* ws = Ws + t * 64;
#pragma unroll
    for (int j = 0; j < 8; ++j) {
      copy16_g2l(arow + j * 8, as + j * 8);
      copy16_g2l(wrow + j * 8, ws + j * 8);
    }
  }
  wait_async_copies();
  __syncthreads();

  for (int k0 = 0; k0 < DMODEL; k0 += 64) {
    const int buf = (k0 >> 6) & 1;
    const bf16_t* curA = As + buf * (128 * 64);
    const bf16_t* curW = Ws + buf * (128 * 64);
    bf16_t* nxtA = As + (buf ^ 1) * (128 * 64) + t * 64;
    bf16_t* nxtW = Ws + (buf ^ 1) * (128 * 64) + t * 64;
    const bool more = (k0 + 64) < DMODEL;

    // kick off next k-step's global->LDS copies (hidden behind the WMMAs)
    if (more) {
      const bf16_t* ap = arow + k0 + 64;
      const bf16_t* wp = wrow + k0 + 64;
#pragma unroll
      for (int j = 0; j < 8; ++j) {
        copy16_g2l(ap + j * 8, nxtA + j * 8);
        copy16_g2l(wp + j * 8, nxtW + j * 8);
      }
    }

    // compute on current buffer: 2 k-chunks of 32, 16 WMMA each
#pragma unroll
    for (int kc = 0; kc < 2; ++kc) {
      v16bf af[4], bf[4];
#pragma unroll
      for (int mt = 0; mt < 4; ++mt) {
        const bf16_t* p = curA + (wm * 64 + mt * 16 + ln) * 64 + kc * 32 + koff;
        af[mt] = mk16(*(const v8bf*)p, *(const v8bf*)(p + 16));
      }
#pragma unroll
      for (int nt = 0; nt < 4; ++nt) {
        const bf16_t* p = curW + (wn * 64 + nt * 16 + ln) * 64 + kc * 32 + koff;
        bf[nt] = mk16(*(const v8bf*)p, *(const v8bf*)(p + 16));
      }
#pragma unroll
      for (int mt = 0; mt < 4; ++mt)
#pragma unroll
        for (int nt = 0; nt < 4; ++nt)
          acc[mt][nt] = wmma_bf16(af[mt], bf[nt], acc[mt][nt]);
    }

    wait_async_copies();  // next buffer fully resident in LDS
    __syncthreads();
  }

  // epilogue
#pragma unroll
  for (int mt = 0; mt < 4; ++mt) {
#pragma unroll
    for (int nt = 0; nt < 4; ++nt) {
      const int rg0 = mBase + wm * 64 + mt * 16 + half * 8;  // first of 8 rows
      const int cg  = nBase + wn * 64 + nt * 16 + ln;
      const float bv = bias[cg];
      if (mode == 1) {
        // Vt[b,h,d,s]: the 8 rows are contiguous in s -> one 16B store
        const int bb = rg0 >> 11, s0 = rg0 & (SEQ - 1);
        const int hh = cg >> 6, dd = cg & 63;
        bf16_t* dst = outB + (((size_t)bb * HEADS + hh) * DHEAD + dd) * SEQ + s0;
        v8bf pk;
#pragma unroll
        for (int i = 0; i < 8; ++i) pk[i] = (bf16_t)(acc[mt][nt][i] + bv);
        *(v8bf*)dst = pk;
      } else if (mode == 0) {
#pragma unroll
        for (int i = 0; i < 8; ++i)
          outB[(size_t)(rg0 + i) * DMODEL + cg] = (bf16_t)(acc[mt][nt][i] + bv);
      } else {
#pragma unroll
        for (int i = 0; i < 8; ++i)
          outF[(size_t)(rg0 + i) * DMODEL + cg] = acc[mt][nt][i] + bv;
      }
    }
  }
}

// ---------------------------------------------------------------------------
// Flash attention: grid (S/64, B*H), block 64 threads = 2 waves.
// Each wave owns 32 q rows (2 m-tiles), iterates causal kv tiles of 64.
// Q,K: [B,S,D] bf16 (Dh-contiguous). Vt: [B,H,Dh,S] bf16. O: [B,S,D] bf16.
// ---------------------------------------------------------------------------
__global__ __launch_bounds__(64) void attn_kernel(const bf16_t* __restrict__ Q,
                                                  const bf16_t* __restrict__ K,
                                                  const bf16_t* __restrict__ Vt,
                                                  const int* __restrict__ msk,
                                                  bf16_t* __restrict__ O) {
  __shared__ bf16_t Pl[2][32 * 64];  // per-wave P tile (rows x kv-cols)

  const int t    = threadIdx.x;
  const int lane = t & 31;
  const int wave = t >> 5;
  const int ln   = lane & 15;
  const int half = lane >> 4;
  const int koff = half * 8;
  const int qb = blockIdx.x * 64 + wave * 32;  // this wave's first q row
  const int bh = blockIdx.y;
  const int bb = bh >> 4, hh = bh & 15;

  const bf16_t* Qp = Q + (size_t)bb * SEQ * DMODEL + hh * DHEAD;
  const bf16_t* Kp = K + (size_t)bb * SEQ * DMODEL + hh * DHEAD;
  const bf16_t* Vp = Vt + ((size_t)bb * HEADS + hh) * DHEAD * SEQ;
  const int* mp = msk + bb * SEQ;

  // Q fragments (persistent): [m-tile][k-chunk of Dh]
  v16bf qf[2][2];
#pragma unroll
  for (int mt = 0; mt < 2; ++mt) {
    const bf16_t* rp = Qp + (size_t)(qb + mt * 16 + ln) * DMODEL;
#pragma unroll
    for (int kc = 0; kc < 2; ++kc) {
      const bf16_t* p = rp + kc * 32 + koff;
      qf[mt][kc] = mk16(*(const v8bf*)p, *(const v8bf*)(p + 16));
    }
  }

  v8f o[2][4];
#pragma unroll
  for (int mt = 0; mt < 2; ++mt)
#pragma unroll
    for (int dt = 0; dt < 4; ++dt) o[mt][dt] = (v8f)0.0f;
  float mrun[2][8], lrun[2][8];
#pragma unroll
  for (int mt = 0; mt < 2; ++mt)
#pragma unroll
    for (int i = 0; i < 8; ++i) { mrun[mt][i] = -1e30f; lrun[mt][i] = 0.0f; }

  for (int kv = 0; kv < qb + 32; kv += 64) {
    // K fragments [n-tile][k-chunk], V fragments [d-tile][kv-chunk], pad mask
    v16bf kf[4][2], vf[4][2];
    int mv[4];
#pragma unroll
    for (int nt = 0; nt < 4; ++nt) {
      const int row = kv + nt * 16 + ln;
      const bf16_t* rp = Kp + (size_t)row * DMODEL;
#pragma unroll
      for (int kc = 0; kc < 2; ++kc) {
        const bf16_t* p = rp + kc * 32 + koff;
        kf[nt][kc] = mk16(*(const v8bf*)p, *(const v8bf*)(p + 16));
      }
      mv[nt] = mp[row];
    }
#pragma unroll
    for (int dt = 0; dt < 4; ++dt) {
      const bf16_t* rp = Vp + (size_t)(dt * 16 + ln) * SEQ + kv;
#pragma unroll
      for (int kc = 0; kc < 2; ++kc) {
        const bf16_t* p = rp + kc * 32 + koff;
        vf[dt][kc] = mk16(*(const v8bf*)p, *(const v8bf*)(p + 16));
      }
    }

    // prefetch next kv tile (global_prefetch_b8): one line covers 128B row chunk
    const int kvn = kv + 64;
    if (kvn < qb + 32) {
#pragma unroll
      for (int nt = 0; nt < 4; ++nt)
        __builtin_prefetch(Kp + (size_t)(kvn + nt * 16 + ln) * DMODEL, 0, 3);
#pragma unroll
      for (int dt = 0; dt < 4; ++dt)
        __builtin_prefetch(Vp + (size_t)(dt * 16 + ln) * SEQ + kvn, 0, 3);
    }

    // scores = Q @ K^T
    v8f sc[2][4];
#pragma unroll
    for (int mt = 0; mt < 2; ++mt)
#pragma unroll
      for (int nt = 0; nt < 4; ++nt) {
        sc[mt][nt] = (v8f)0.0f;
#pragma unroll
        for (int kc = 0; kc < 2; ++kc)
          sc[mt][nt] = wmma_bf16(qf[mt][kc], kf[nt][kc], sc[mt][nt]);
      }

    // scale + causal + pad mask
#pragma unroll
    for (int mt = 0; mt < 2; ++mt)
#pragma unroll
      for (int nt = 0; nt < 4; ++nt) {
        const int col = kv + nt * 16 + ln;
#pragma unroll
        for (int i = 0; i < 8; ++i) {
          const int row = qb + mt * 16 + half * 8 + i;
          float v = sc[mt][nt][i] * 0.125f;
          if (col > row || mv[nt] == 0) v = -1e30f;
          sc[mt][nt][i] = v;
        }
      }

    // online softmax
#pragma unroll
    for (int mt = 0; mt < 2; ++mt) {
      float alpha[8];
#pragma unroll
      for (int i = 0; i < 8; ++i) {
        float mx = sc[mt][0][i];
#pragma unroll
        for (int nt = 1; nt < 4; ++nt) mx = fmaxf(mx, sc[mt][nt][i]);
#pragma unroll
        for (int off = 1; off < 16; off <<= 1)
          mx = fmaxf(mx, __shfl_xor(mx, off, 32));  // stays within 16-lane half
        const float mn = fmaxf(mrun[mt][i], mx);
        alpha[i] = __expf(mrun[mt][i] - mn);
        mrun[mt][i] = mn;
        float rs = 0.0f;
#pragma unroll
        for (int nt = 0; nt < 4; ++nt) {
          const float e = __expf(sc[mt][nt][i] - mn);
          sc[mt][nt][i] = e;
          rs += e;
        }
#pragma unroll
        for (int off = 1; off < 16; off <<= 1) rs += __shfl_xor(rs, off, 32);
        lrun[mt][i] = lrun[mt][i] * alpha[i] + rs;
      }
#pragma unroll
      for (int dt = 0; dt < 4; ++dt)
#pragma unroll
        for (int i = 0; i < 8; ++i) o[mt][dt][i] *= alpha[i];
    }

    // P (C-layout) -> LDS row-major [32 rows][64 kv] per wave
#pragma unroll
    for (int mt = 0; mt < 2; ++mt)
#pragma unroll
      for (int nt = 0; nt < 4; ++nt)
#pragma unroll
        for (int i = 0; i < 8; ++i)
          Pl[wave][(mt * 16 + half * 8 + i) * 64 + nt * 16 + ln] =
              (bf16_t)sc[mt][nt][i];

    // re-read P as A-fragments (same-wave LDS ops are in-order)
    v16bf pf[2][2];
#pragma unroll
    for (int mt = 0; mt < 2; ++mt) {
      const bf16_t* rp = &Pl[wave][(mt * 16 + ln) * 64];
#pragma unroll
      for (int kc = 0; kc < 2; ++kc) {
        const bf16_t* p = rp + kc * 32 + koff;
        pf[mt][kc] = mk16(*(const v8bf*)p, *(const v8bf*)(p + 16));
      }
    }

    // O += P @ V
#pragma unroll
    for (int mt = 0; mt < 2; ++mt)
#pragma unroll
      for (int dt = 0; dt < 4; ++dt)
#pragma unroll
        for (int kc = 0; kc < 2; ++kc)
          o[mt][dt] = wmma_bf16(pf[mt][kc], vf[dt][kc], o[mt][dt]);
  }

  // epilogue: normalize rows and store O[B,S,D] bf16
#pragma unroll
  for (int mt = 0; mt < 2; ++mt)
#pragma unroll
    for (int i = 0; i < 8; ++i) {
      const float inv = 1.0f / lrun[mt][i];
      const int row = qb + mt * 16 + half * 8 + i;
#pragma unroll
      for (int dt = 0; dt < 4; ++dt) {
        const int col = hh * DHEAD + dt * 16 + ln;
        O[((size_t)bb * SEQ + row) * DMODEL + col] = (bf16_t)(o[mt][dt][i] * inv);
      }
    }
}

// ---------------------------------------------------------------------------
extern "C" void kernel_launch(void* const* d_in, const int* in_sizes, int n_in,
                              void* d_out, int out_size, void* d_ws, size_t ws_size,
                              hipStream_t stream) {
  (void)in_sizes; (void)n_in; (void)out_size; (void)ws_size;
  const float* x  = (const float*)d_in[0];
  const int* mask = (const int*)d_in[1];
  const float* Wq = (const float*)d_in[2];
  const float* bq = (const float*)d_in[3];
  const float* Wk = (const float*)d_in[4];
  const float* bk = (const float*)d_in[5];
  const float* Wv = (const float*)d_in[6];
  const float* bv = (const float*)d_in[7];
  const float* Wo = (const float*)d_in[8];
  const float* bo = (const float*)d_in[9];

  const size_t XN = (size_t)MTOT * DMODEL;   // 8388608
  const size_t WN = (size_t)DMODEL * DMODEL; // 1048576

  bf16_t* xb  = (bf16_t*)d_ws;
  bf16_t* wqb = xb + XN;
  bf16_t* wkb = wqb + WN;
  bf16_t* wvb = wkb + WN;
  bf16_t* wob = wvb + WN;
  bf16_t* Qb  = wob + WN;
  bf16_t* Kb  = Qb + XN;
  bf16_t* Vtb = Kb + XN;
  bf16_t* Ob  = Vtb + XN;

  cvt_bf16_kernel<<<2048, 256, 0, stream>>>(x, xb, (int)(XN / 4));
  cvt_bf16_kernel<<<512, 256, 0, stream>>>(Wq, wqb, (int)(WN / 4));
  cvt_bf16_kernel<<<512, 256, 0, stream>>>(Wk, wkb, (int)(WN / 4));
  cvt_bf16_kernel<<<512, 256, 0, stream>>>(Wv, wvb, (int)(WN / 4));
  cvt_bf16_kernel<<<512, 256, 0, stream>>>(Wo, wob, (int)(WN / 4));

  dim3 ggrid(DMODEL / 128, MTOT / 128);  // (8, 64)
  gemm_kernel<<<ggrid, 128, 0, stream>>>(xb, wqb, bq, Qb, nullptr, 0);
  gemm_kernel<<<ggrid, 128, 0, stream>>>(xb, wkb, bk, Kb, nullptr, 0);
  gemm_kernel<<<ggrid, 128, 0, stream>>>(xb, wvb, bv, Vtb, nullptr, 1);

  attn_kernel<<<dim3(SEQ / 64, BATCH * HEADS), 64, 0, stream>>>(Qb, Kb, Vtb, mask, Ob);

  gemm_kernel<<<ggrid, 128, 0, stream>>>(Ob, wob, bo, nullptr, (float*)d_out, 2);
}
